// TransformerLayerWithCross_72146860638707
// MI455X (gfx1250) — matrix operand order repaired and checked by reading, not compiled, not run on hardware
//
#include <hip/hip_runtime.h>
#include <hip/hip_bf16.h>

typedef __attribute__((ext_vector_type(16))) _Float16 v16h;
typedef __attribute__((ext_vector_type(8)))  float    v8f;

struct alignas(16) U4 { unsigned int x, y, z, w; };
struct alignas(16) F4 { float x, y, z, w; };
union Frag16 { v16h h; U4 u[2]; };

static constexpr int kB  = 2;
static constexpr int kLq = 2048;
static constexpr int kLo = 1024;
static constexpr int kD  = 1024;
static constexpr int kH  = 16;
static constexpr int kHD = 64;
static constexpr int kFF = 4096;

__device__ __forceinline__ v8f wmma16(v16h a, v16h b, v8f c) {
  return __builtin_amdgcn_wmma_f32_16x16x32_f16(false, a, false, b, (short)0, c, false, false);
}

// ---------------------------------------------------------------------------
// CDNA5 async global->LDS path (guarded: falls back to VGPR staging if absent)
// Builtin signature (from hipcc diagnostic): (int4 AS1*, int4 AS3*, imm, imm)
// ---------------------------------------------------------------------------
#if defined(__gfx1250__) && __has_builtin(__builtin_amdgcn_global_load_async_to_lds_b128)
#define HAVE_ASYNC_LDS 1
typedef int i4v __attribute__((vector_size(4 * sizeof(int))));
typedef __attribute__((address_space(1))) i4v g1_i4;
typedef __attribute__((address_space(3))) i4v l3_i4;
__device__ __forceinline__ void async_cp16(void* lds_dst, const void* gsrc) {
  __builtin_amdgcn_global_load_async_to_lds_b128((g1_i4*)gsrc, (l3_i4*)lds_dst, 0, 0);
}
#else
#define HAVE_ASYNC_LDS 0
#endif

__device__ __forceinline__ void wait_async0() {
#if defined(__gfx1250__) && __has_builtin(__builtin_amdgcn_s_wait_asynccnt)
  __builtin_amdgcn_s_wait_asynccnt(0);
#elif defined(__gfx1250__)
  asm volatile("s_wait_asynccnt 0" ::: "memory");
#endif
}

// width-16 xor-butterfly step: VALU permlane16 if available, else shfl (ds unit)
__device__ __forceinline__ float xor16_step(float v, int mask,
                                            unsigned sel0, unsigned sel1) {
#if defined(__gfx1250__) && __has_builtin(__builtin_amdgcn_permlane16)
  unsigned u = __builtin_bit_cast(unsigned, v);
  u = __builtin_amdgcn_permlane16(u, u, sel0, sel1, false, false);
  return __builtin_bit_cast(float, u);
#else
  return __shfl_xor(v, mask, 32);
#endif
}

__device__ __forceinline__ float rowmax16(float t) {
  t = fmaxf(t, xor16_step(t, 1, 0x67452301u, 0xEFCDAB89u));
  t = fmaxf(t, xor16_step(t, 2, 0x45670123u, 0xCDEF89ABu));
  t = fmaxf(t, xor16_step(t, 4, 0x32107654u, 0xBA98FEDCu));
  t = fmaxf(t, xor16_step(t, 8, 0xFEDCBA98u, 0x76543210u));
  return t;
}

// ---------------------------------------------------------------------------
// Weight cast + transpose: W[K,N] f32 -> Wt[N,K] f16  (LDS 32x32 tile)
// ---------------------------------------------------------------------------
__global__ __launch_bounds__(256) void wcast_t_kernel(
    const float* __restrict__ W, _Float16* __restrict__ Wt, int Kd, int Nd) {
  __shared__ float t[32][33];
  const int n0 = blockIdx.x * 32, k0 = blockIdx.y * 32;
  const int tx = threadIdx.x, ty = threadIdx.y;
#pragma unroll
  for (int i = 0; i < 4; ++i)
    t[ty + 8 * i][tx] = W[(size_t)(k0 + ty + 8 * i) * Nd + n0 + tx];
  __syncthreads();
#pragma unroll
  for (int i = 0; i < 4; ++i)
    Wt[(size_t)(n0 + ty + 8 * i) * Kd + k0 + tx] = (_Float16)t[tx][ty + 8 * i];
}

// ---------------------------------------------------------------------------
// Row LayerNorm over D=1024, f32 in -> f16 out. One block (256 thr) per row.
// ---------------------------------------------------------------------------
__global__ __launch_bounds__(256) void layernorm_f16_kernel(
    const float* __restrict__ X, const float* __restrict__ gw,
    const float* __restrict__ gb, _Float16* __restrict__ Y) {
  __shared__ float red[256];
  const int tid = threadIdx.x;
  const size_t rbase = (size_t)blockIdx.x * kD;
  F4 xv = *(const F4*)&X[rbase + tid * 4];
  red[tid] = xv.x + xv.y + xv.z + xv.w;
  __syncthreads();
  for (int o = 128; o > 0; o >>= 1) {
    if (tid < o) red[tid] += red[tid + o];
    __syncthreads();
  }
  const float mu = red[0] * (1.0f / kD);
  __syncthreads();
  red[tid] = xv.x * xv.x + xv.y * xv.y + xv.z * xv.z + xv.w * xv.w;
  __syncthreads();
  for (int o = 128; o > 0; o >>= 1) {
    if (tid < o) red[tid] += red[tid + o];
    __syncthreads();
  }
  const float var = red[0] * (1.0f / kD) - mu * mu;
  const float rstd = rsqrtf(var + 1e-5f);
  const int c = tid * 4;
  float v0 = (xv.x - mu) * rstd * gw[c + 0] + gb[c + 0];
  float v1 = (xv.y - mu) * rstd * gw[c + 1] + gb[c + 1];
  float v2 = (xv.z - mu) * rstd * gw[c + 2] + gb[c + 2];
  float v3 = (xv.w - mu) * rstd * gw[c + 3] + gb[c + 3];
  Y[rbase + c + 0] = (_Float16)v0;
  Y[rbase + c + 1] = (_Float16)v1;
  Y[rbase + c + 2] = (_Float16)v2;
  Y[rbase + c + 3] = (_Float16)v3;
}

// ---------------------------------------------------------------------------
// Plain cast f32 -> f16 (for `other`, no LN)
// ---------------------------------------------------------------------------
__global__ __launch_bounds__(256) void cast_f16_kernel(
    const float* __restrict__ X, _Float16* __restrict__ Y, long n) {
  long i = ((long)blockIdx.x * 256 + threadIdx.x) * 4;
  if (i + 3 < n) {
    F4 v = *(const F4*)&X[i];
    Y[i + 0] = (_Float16)v.x;
    Y[i + 1] = (_Float16)v.y;
    Y[i + 2] = (_Float16)v.z;
    Y[i + 3] = (_Float16)v.w;
  }
}

// ---------------------------------------------------------------------------
// WMMA GEMM: C[M,N] = A[M,K](f16, row-major) @ Wt[N,K](f16)^T + bias
// Tile 128x128, BK=32, 256 threads = 8 waves, each wave 4(M)x2(N) fragments.
// Epilogue MODE: 1 = f16 per-head [b,h,m,d]   2 = f16 transposed-V [b,h,d,m]
//               3 = f32 + residual            4 = f16 exact GELU
// ---------------------------------------------------------------------------
template <int MODE>
__global__ __launch_bounds__(256) void gemm_wmma_kernel(
    const _Float16* __restrict__ A, const _Float16* __restrict__ Bt,
    const float* __restrict__ bias, const float* __restrict__ resid,
    float* __restrict__ outf, _Float16* __restrict__ outh,
    int M, int N, int K, int Ltok) {
  constexpr int LDA = 40;  // 32 halves + 8 pad (keeps 16B alignment)
  __shared__ _Float16 sA[128 * LDA];
  __shared__ _Float16 sB[128 * LDA];
  const int tid = threadIdx.x;
  const int lane = tid & 31;
  const int w = tid >> 5;
  const int hi = lane >> 4;
  const int l15 = lane & 15;
  const int wMb = (w >> 2) * 64;
  const int wNb = (w & 3) * 32;
  const int bx = blockIdx.x, by = blockIdx.y;

  v8f acc[4][2];
#pragma unroll
  for (int i = 0; i < 4; ++i)
#pragma unroll
    for (int j = 0; j < 2; ++j)
      acc[i][j] = (v8f){0.f, 0.f, 0.f, 0.f, 0.f, 0.f, 0.f, 0.f};

  for (int kb = 0; kb < K; kb += 32) {
    const _Float16* Ag = A + (size_t)(by * 128) * K + kb;
    const _Float16* Bg = Bt + (size_t)(bx * 128) * K + kb;
#if HAVE_ASYNC_LDS
#pragma unroll
    for (int i = 0; i < 2; ++i) {
      const int c = tid + 256 * i;
      const int r = c >> 2;
      const int cc = (c & 3) << 3;
      async_cp16(&sA[r * LDA + cc], &Ag[(size_t)r * K + cc]);
      async_cp16(&sB[r * LDA + cc], &Bg[(size_t)r * K + cc]);
    }
    wait_async0();
    __syncthreads();
#else
#pragma unroll
    for (int i = 0; i < 2; ++i) {
      const int c = tid + 256 * i;
      const int r = c >> 2;
      const int cc = (c & 3) << 3;
      *(U4*)&sA[r * LDA + cc] = *(const U4*)&Ag[(size_t)r * K + cc];
      *(U4*)&sB[r * LDA + cc] = *(const U4*)&Bg[(size_t)r * K + cc];
    }
    __syncthreads();
#endif

    Frag16 af[4], bf[2];
#pragma unroll
    for (int mi = 0; mi < 4; ++mi) {
      const int base = (wMb + mi * 16 + l15) * LDA;
      af[mi].u[0] = *(const U4*)&sA[base + hi * 8];       // halves 0..7 : K=hi*8..
      af[mi].u[1] = *(const U4*)&sA[base + 16 + hi * 8];  // halves 8..15: K=16+hi*8..
    }
#pragma unroll
    for (int ni = 0; ni < 2; ++ni) {
      const int base = (wNb + ni * 16 + l15) * LDA + hi * 8;  // lane=N, halves=K
      bf[ni].u[0] = *(const U4*)&sB[base];
      bf[ni].u[1] = *(const U4*)&sB[base + 16];
    }
#pragma unroll
    for (int mi = 0; mi < 4; ++mi)
#pragma unroll
      for (int ni = 0; ni < 2; ++ni)
        acc[mi][ni] = wmma16(af[mi].h, bf[ni].h, acc[mi][ni]);
    __syncthreads();
  }

  const int m0 = by * 128 + wMb;
  const int n0 = bx * 128 + wNb;
#pragma unroll
  for (int mi = 0; mi < 4; ++mi) {
#pragma unroll
    for (int ni = 0; ni < 2; ++ni) {
#pragma unroll
      for (int r = 0; r < 8; ++r) {
        const int row = m0 + mi * 16 + r + hi * 8;  // C layout: VGPR r -> M=r(+8)
        const int col = n0 + ni * 16 + l15;
        float v = acc[mi][ni][r] + bias[col];
        if (MODE == 1) {  // Q/K per-head: [b,h,m,hd]
          const int b = row / Ltok, m = row % Ltok;
          const int h = col >> 6, d = col & 63;
          outh[((((size_t)b * kH + h) * Ltok + m) << 6) + d] = (_Float16)v;
        } else if (MODE == 2) {  // V transposed: [b,h,hd,m]
          const int b = row / Ltok, m = row % Ltok;
          const int h = col >> 6, d = col & 63;
          outh[(((size_t)b * kH + h) * kHD + d) * (size_t)Ltok + m] = (_Float16)v;
        } else if (MODE == 3) {  // f32 + residual
          const size_t idx = (size_t)row * N + col;
          outf[idx] = v + resid[idx];
        } else {  // MODE == 4: exact GELU, f16 out
          v = 0.5f * v * (1.0f + erff(v * 0.70710678118654752f));
          outh[(size_t)row * N + col] = (_Float16)v;
        }
      }
    }
  }
}

// ---------------------------------------------------------------------------
// Flash attention v2: per wave 16 queries x hd=64, keys in chunks of 64.
// Row-sums computed by an extra "ones" WMMA accumulator (no shuffle needed);
// only the row-max uses a width-16 butterfly.
// Q:[bh,Lq,64] f16, K:[bh,Lk,64] f16, Vt:[bh,64,Lk] f16 -> Out f16 [b*Lq, D]
// ---------------------------------------------------------------------------
__global__ __launch_bounds__(256) void flash_attn_kernel(
    const _Float16* __restrict__ Q, const _Float16* __restrict__ Km,
    const _Float16* __restrict__ Vt, _Float16* __restrict__ Out,
    int Lq, int Lk, float scale) {
  constexpr int LDP = 72;  // 64 + 8 pad halves (keeps 16B alignment)
  __shared__ _Float16 sP[8 * 16 * LDP];
  const int tid = threadIdx.x;
  const int lane = tid & 31, w = tid >> 5;
  const int hi = lane >> 4, l15 = lane & 15;
  const int bh = blockIdx.y;
  const int b = bh >> 4, h = bh & 15;
  const int qm0 = blockIdx.x * 128 + w * 16;
  const _Float16* qb = Q + ((size_t)bh * Lq + qm0) * kHD;
  const _Float16* kb = Km + (size_t)bh * Lk * kHD;
  const _Float16* vb = Vt + (size_t)bh * kHD * Lk;
  _Float16* sp = &sP[w * 16 * LDP];

  // Q A-fragments (K-dim = head dim, 2 chunks of 32)
  Frag16 qa[2];
  {
    const _Float16* qp = qb + (size_t)l15 * kHD;
#pragma unroll
    for (int c = 0; c < 2; ++c) {
      qa[c].u[0] = *(const U4*)(qp + c * 32 + hi * 8);
      qa[c].u[1] = *(const U4*)(qp + c * 32 + 16 + hi * 8);
    }
  }
  Frag16 onesf;
#pragma unroll
  for (int i = 0; i < 16; ++i) onesf.h[i] = (_Float16)1.0f;

  v8f o[4], o4;
  float mrow[8];
#pragma unroll
  for (int j = 0; j < 4; ++j) o[j] = (v8f){0.f, 0.f, 0.f, 0.f, 0.f, 0.f, 0.f, 0.f};
  o4 = (v8f){0.f, 0.f, 0.f, 0.f, 0.f, 0.f, 0.f, 0.f};
#pragma unroll
  for (int i = 0; i < 8; ++i) mrow[i] = -3.0e38f;

  for (int kt = 0; kt < Lk; kt += 64) {
    // ---- S = Q @ K^T for 64 keys (four 16-key N-tiles) ----
    v8f s[4];
#pragma unroll
    for (int j = 0; j < 4; ++j) {
      s[j] = (v8f){0.f, 0.f, 0.f, 0.f, 0.f, 0.f, 0.f, 0.f};
      const _Float16* kp = kb + (size_t)(kt + j * 16 + l15) * kHD + hi * 8;
      Frag16 f0, f1;
      f0.u[0] = *(const U4*)(kp);      f0.u[1] = *(const U4*)(kp + 16);
      f1.u[0] = *(const U4*)(kp + 32); f1.u[1] = *(const U4*)(kp + 48);
      s[j] = wmma16(qa[0].h, f0.h, s[j]);
      s[j] = wmma16(qa[1].h, f1.h, s[j]);
    }
    // ---- online softmax: only the row-max needs a lane butterfly ----
#pragma unroll
    for (int i = 0; i < 8; ++i) {
      const float a0 = s[0][i] * scale;
      const float a1 = s[1][i] * scale;
      const float a2 = s[2][i] * scale;
      const float a3 = s[3][i] * scale;
      float t = fmaxf(fmaxf(a0, a1), fmaxf(a2, a3));
      t = rowmax16(t);
      const float mn = fmaxf(mrow[i], t);
      const float corr = expf(mrow[i] - mn);
      mrow[i] = mn;
      const float p0 = expf(a0 - mn);
      const float p1 = expf(a1 - mn);
      const float p2 = expf(a2 - mn);
      const float p3 = expf(a3 - mn);
      o[0][i] *= corr; o[1][i] *= corr; o[2][i] *= corr; o[3][i] *= corr;
      o4[i] *= corr;   // running row-sum rescales identically
      _Float16* pr = sp + (i + hi * 8) * LDP + l15;
      pr[0]  = (_Float16)p0;
      pr[16] = (_Float16)p1;
      pr[32] = (_Float16)p2;
      pr[48] = (_Float16)p3;
    }
    __syncthreads();
    // ---- reload P as A-fragments (16x64 -> 2 chunks of K=32) ----
    Frag16 pf[2];
#pragma unroll
    for (int c = 0; c < 2; ++c) {
      pf[c].u[0] = *(const U4*)&sp[l15 * LDP + c * 32 + hi * 8];
      pf[c].u[1] = *(const U4*)&sp[l15 * LDP + c * 32 + 16 + hi * 8];
    }
    // row sums via matrix unit: l += P @ ones
    o4 = wmma16(pf[0].h, onesf.h, o4);
    o4 = wmma16(pf[1].h, onesf.h, o4);
    // ---- O += P @ V ----
#pragma unroll
    for (int j = 0; j < 4; ++j) {
#pragma unroll
      for (int c = 0; c < 2; ++c) {
        Frag16 vf;
        const _Float16* vp = vb + (size_t)(j * 16 + l15) * Lk + kt + c * 32 + hi * 8;
        vf.u[0] = *(const U4*)(vp);
        vf.u[1] = *(const U4*)(vp + 16);
        o[j] = wmma16(pf[c].h, vf.h, o[j]);
      }
    }
    __syncthreads();
  }

#pragma unroll
  for (int j = 0; j < 4; ++j)
#pragma unroll
    for (int i = 0; i < 8; ++i) {
      const int row = qm0 + i + hi * 8;
      const float v = o[j][i] / o4[i];
      Out[((size_t)b * Lq + row) * kD + h * kHD + j * 16 + l15] = (_Float16)v;
    }
}

// ---------------------------------------------------------------------------
extern "C" void kernel_launch(void* const* d_in, const int* in_sizes, int n_in,
                              void* d_out, int out_size, void* d_ws, size_t ws_size,
                              hipStream_t stream) {
  (void)in_sizes; (void)n_in; (void)out_size; (void)ws_size;
  const float* x      = (const float*)d_in[0];
  const float* other  = (const float*)d_in[1];
  const float* sa_wq = (const float*)d_in[2];  const float* sa_bq = (const float*)d_in[3];
  const float* sa_wk = (const float*)d_in[4];  const float* sa_bk = (const float*)d_in[5];
  const float* sa_wv = (const float*)d_in[6];  const float* sa_bv = (const float*)d_in[7];
  const float* sa_wo = (const float*)d_in[8];  const float* sa_bo = (const float*)d_in[9];
  const float* ca_wq = (const float*)d_in[10]; const float* ca_bq = (const float*)d_in[11];
  const float* ca_wk = (const float*)d_in[12]; const float* ca_bk = (const float*)d_in[13];
  const float* ca_wv = (const float*)d_in[14]; const float* ca_bv = (const float*)d_in[15];
  const float* ca_wo = (const float*)d_in[16]; const float* ca_bo = (const float*)d_in[17];
  const float* ln1w = (const float*)d_in[18];  const float* ln1b = (const float*)d_in[19];
  const float* ln2w = (const float*)d_in[20];  const float* ln2b = (const float*)d_in[21];
  const float* ln3w = (const float*)d_in[22];  const float* ln3b = (const float*)d_in[23];
  const float* fc1w = (const float*)d_in[24];  const float* fc1b = (const float*)d_in[25];
  const float* fc2w = (const float*)d_in[26];  const float* fc2b = (const float*)d_in[27];

  const int Mq = kB * kLq;  // 4096 query tokens
  const int Mo = kB * kLo;  // 2048 kv tokens (cross)

  char* p = (char*)d_ws;
  auto alloc = [&](size_t bytes) -> void* {
    void* r = (void*)p;
    p += (bytes + 255) & ~(size_t)255;
    return r;
  };
  _Float16* wt_saq = (_Float16*)alloc((size_t)kD * kD * 2);
  _Float16* wt_sak = (_Float16*)alloc((size_t)kD * kD * 2);
  _Float16* wt_sav = (_Float16*)alloc((size_t)kD * kD * 2);
  _Float16* wt_sao = (_Float16*)alloc((size_t)kD * kD * 2);
  _Float16* wt_caq = (_Float16*)alloc((size_t)kD * kD * 2);
  _Float16* wt_cak = (_Float16*)alloc((size_t)kD * kD * 2);
  _Float16* wt_cav = (_Float16*)alloc((size_t)kD * kD * 2);
  _Float16* wt_cao = (_Float16*)alloc((size_t)kD * kD * 2);
  _Float16* wt_fc1 = (_Float16*)alloc((size_t)kD * kFF * 2);
  _Float16* wt_fc2 = (_Float16*)alloc((size_t)kD * kFF * 2);
  _Float16* xln    = (_Float16*)alloc((size_t)Mq * kD * 2);
  _Float16* oth16  = (_Float16*)alloc((size_t)Mo * kD * 2);
  _Float16* Qb     = (_Float16*)alloc((size_t)Mq * kD * 2);
  _Float16* Kb     = (_Float16*)alloc((size_t)Mq * kD * 2);
  _Float16* Vtb    = (_Float16*)alloc((size_t)Mq * kD * 2);
  _Float16* attn   = (_Float16*)alloc((size_t)Mq * kD * 2);
  float*    x1     = (float*)alloc((size_t)Mq * kD * 4);
  float*    x2     = (float*)alloc((size_t)Mq * kD * 4);
  _Float16* h1     = (_Float16*)alloc((size_t)Mq * kFF * 2);

  const dim3 tb(32, 8);
  const dim3 gDD(kD / 32, kD / 32);
  wcast_t_kernel<<<gDD, tb, 0, stream>>>(sa_wq, wt_saq, kD, kD);
  wcast_t_kernel<<<gDD, tb, 0, stream>>>(sa_wk, wt_sak, kD, kD);
  wcast_t_kernel<<<gDD, tb, 0, stream>>>(sa_wv, wt_sav, kD, kD);
  wcast_t_kernel<<<gDD, tb, 0, stream>>>(sa_wo, wt_sao, kD, kD);
  wcast_t_kernel<<<gDD, tb, 0, stream>>>(ca_wq, wt_caq, kD, kD);
  wcast_t_kernel<<<gDD, tb, 0, stream>>>(ca_wk, wt_cak, kD, kD);
  wcast_t_kernel<<<gDD, tb, 0, stream>>>(ca_wv, wt_cav, kD, kD);
  wcast_t_kernel<<<gDD, tb, 0, stream>>>(ca_wo, wt_cao, kD, kD);
  wcast_t_kernel<<<dim3(kFF / 32, kD / 32), tb, 0, stream>>>(fc1w, wt_fc1, kD, kFF);
  wcast_t_kernel<<<dim3(kD / 32, kFF / 32), tb, 0, stream>>>(fc2w, wt_fc2, kFF, kD);

  // ---- self-attention ----
  layernorm_f16_kernel<<<Mq, 256, 0, stream>>>(x, ln1w, ln1b, xln);
  const dim3 gPrj(kD / 128, Mq / 128);
  gemm_wmma_kernel<1><<<gPrj, 256, 0, stream>>>(xln, wt_saq, sa_bq, nullptr, nullptr, Qb,  Mq, kD, kD, kLq);
  gemm_wmma_kernel<1><<<gPrj, 256, 0, stream>>>(xln, wt_sak, sa_bk, nullptr, nullptr, Kb,  Mq, kD, kD, kLq);
  gemm_wmma_kernel<2><<<gPrj, 256, 0, stream>>>(xln, wt_sav, sa_bv, nullptr, nullptr, Vtb, Mq, kD, kD, kLq);
  flash_attn_kernel<<<dim3(kLq / 128, kB * kH), 256, 0, stream>>>(Qb, Kb, Vtb, attn, kLq, kLq, 0.125f);
  gemm_wmma_kernel<3><<<gPrj, 256, 0, stream>>>(attn, wt_sao, sa_bo, x, x1, nullptr, Mq, kD, kD, kLq);

  // ---- cross-attention (KV from `other`, not normed) ----
  layernorm_f16_kernel<<<Mq, 256, 0, stream>>>(x1, ln2w, ln2b, xln);
  cast_f16_kernel<<<(Mo * kD) / 1024, 256, 0, stream>>>(other, oth16, (long)Mo * kD);
  gemm_wmma_kernel<1><<<gPrj, 256, 0, stream>>>(xln, wt_caq, ca_bq, nullptr, nullptr, Qb, Mq, kD, kD, kLq);
  const dim3 gPrjO(kD / 128, Mo / 128);
  gemm_wmma_kernel<1><<<gPrjO, 256, 0, stream>>>(oth16, wt_cak, ca_bk, nullptr, nullptr, Kb,  Mo, kD, kD, kLo);
  gemm_wmma_kernel<2><<<gPrjO, 256, 0, stream>>>(oth16, wt_cav, ca_bv, nullptr, nullptr, Vtb, Mo, kD, kD, kLo);
  flash_attn_kernel<<<dim3(kLq / 128, kB * kH), 256, 0, stream>>>(Qb, Kb, Vtb, attn, kLq, kLo, 0.125f);
  gemm_wmma_kernel<3><<<gPrj, 256, 0, stream>>>(attn, wt_cao, ca_bo, x1, x2, nullptr, Mq, kD, kD, kLq);

  // ---- FFN ----
  layernorm_f16_kernel<<<Mq, 256, 0, stream>>>(x2, ln3w, ln3b, xln);
  gemm_wmma_kernel<4><<<dim3(kFF / 128, Mq / 128), 256, 0, stream>>>(xln, wt_fc1, fc1b, nullptr, nullptr, h1, Mq, kFF, kD, kLq);
  gemm_wmma_kernel<3><<<gPrj, 256, 0, stream>>>(h1, wt_fc2, fc2b, x2, (float*)d_out, nullptr, Mq, kD, kFF, kLq);
}
